// EncoderBlock_26963804685134
// MI455X (gfx1250) — matrix operand order repaired and checked by reading, not compiled, not run on hardware
//
#include <hip/hip_runtime.h>
#include <hip/hip_bf16.h>
#include <math.h>

// ---------------------------------------------------------------------------
// Transformer encoder block for MI455X (gfx1250, wave32, WMMA).
// B=4, N=2048, D=768, H=12, HD=64, HID=3072.
// All GEMMs use v_wmma_f32_16x16x32_f16 (f16 in, f32 accumulate).
// Double-buffered LDS ping-pong; async global->LDS staging (ASYNCcnt) +
// global_prefetch for L2 warm-up.
// ---------------------------------------------------------------------------

typedef _Float16 f16;
typedef __attribute__((ext_vector_type(16))) _Float16 v16h;
typedef __attribute__((ext_vector_type(8)))  _Float16 v8h;
typedef __attribute__((ext_vector_type(8)))  float    v8f;
typedef int gcc_v4i __attribute__((vector_size(16)));  // matches builtin proto

union F16Frag { v16h v; v8h h[2]; };

#define SEQ_N 2048
#define DIM   768
#define NH    12
#define HDIM  64
#define HID   3072
#define MTOT  (4 * SEQ_N)          // 8192 rows
#define ATT_SCALE 0.125f           // 1/sqrt(64)

#if __has_builtin(__builtin_amdgcn_global_load_async_to_lds_b128) && \
    __has_builtin(__builtin_amdgcn_s_wait_asynccnt)
#define USE_ASYNC 1
#else
#define USE_ASYNC 0
#endif

#if USE_ASYNC
#define ASYNC_CP16(gsrc, ldst)                                            \
  __builtin_amdgcn_global_load_async_to_lds_b128(                         \
      (__attribute__((address_space(1))) gcc_v4i*)(gsrc),                 \
      (__attribute__((address_space(3))) gcc_v4i*)(ldst), 0, 0)
#define ASYNC_WAIT() __builtin_amdgcn_s_wait_asynccnt(0)
#else
#define ASYNC_WAIT()
#endif

static __device__ inline v8f v8f_zero() {
  v8f z;
#pragma unroll
  for (int i = 0; i < 8; ++i) z[i] = 0.0f;
  return z;
}

// ---------------------------------------------------------------------------
// f32 -> f16 elementwise cast (weights)
// ---------------------------------------------------------------------------
__global__ void cvt_f32_to_f16(const float* __restrict__ in, f16* __restrict__ out, int n) {
  int i = blockIdx.x * blockDim.x + threadIdx.x;
  int stride = gridDim.x * blockDim.x;
  for (; i < n; i += stride) out[i] = (f16)in[i];
}

// ---------------------------------------------------------------------------
// LayerNorm over D=768, output f16. One 256-thread block per row.
// ---------------------------------------------------------------------------
__global__ __launch_bounds__(256) void ln_to_f16(const float* __restrict__ x,
                                                 const float* __restrict__ g,
                                                 const float* __restrict__ b,
                                                 f16* __restrict__ out) {
  __shared__ float red[8];
  const int row = blockIdx.x;
  const int tid = threadIdx.x;
  const float* xr = x + (size_t)row * DIM;
  float v0 = xr[tid], v1 = xr[tid + 256], v2 = xr[tid + 512];

  float s = v0 + v1 + v2;
#pragma unroll
  for (int m = 16; m >= 1; m >>= 1) s += __shfl_xor(s, m, 32);
  if ((tid & 31) == 0) red[tid >> 5] = s;
  __syncthreads();
  float tot = 0.0f;
#pragma unroll
  for (int i = 0; i < 8; ++i) tot += red[i];
  const float mu = tot * (1.0f / 768.0f);
  __syncthreads();

  const float d0 = v0 - mu, d1 = v1 - mu, d2 = v2 - mu;
  s = d0 * d0 + d1 * d1 + d2 * d2;
#pragma unroll
  for (int m = 16; m >= 1; m >>= 1) s += __shfl_xor(s, m, 32);
  if ((tid & 31) == 0) red[tid >> 5] = s;
  __syncthreads();
  tot = 0.0f;
#pragma unroll
  for (int i = 0; i < 8; ++i) tot += red[i];
  const float rs = rsqrtf(tot * (1.0f / 768.0f) + 1e-5f);

  f16* orow = out + (size_t)row * DIM;
  orow[tid]       = (f16)(d0 * rs * g[tid]       + b[tid]);
  orow[tid + 256] = (f16)(d1 * rs * g[tid + 256] + b[tid + 256]);
  orow[tid + 512] = (f16)(d2 * rs * g[tid + 512] + b[tid + 512]);
}

// ---------------------------------------------------------------------------
// Tiled WMMA GEMM: C[M,Ncols] = A[M,K] * Bw[K,Ncols]   (A,Bw row-major f16)
// Block tile 128x128, 256 threads = 8 waves (4 along M x 2 along N),
// each wave owns a 32x64 patch = 2x4 grid of 16x16 WMMA tiles, BK=32.
// LDS double-buffered; A staged via async-to-LDS, B transposed via registers.
// ---------------------------------------------------------------------------
constexpr int EPI_QKV = 0;
constexpr int EPI_BIAS_RES = 1;
constexpr int EPI_BIAS_GELU = 2;

template <int EPI>
__global__ __launch_bounds__(256) void gemm_f16(
    const f16* __restrict__ A, const f16* __restrict__ Bw,
    int M, int Ncols, int K,
    const float* __restrict__ bias, const float* __restrict__ resid,
    float* __restrict__ outf, f16* __restrict__ outh,
    f16* __restrict__ qd, f16* __restrict__ kd, f16* __restrict__ vd) {
  constexpr int LDK = 40;  // padded K-stride (elements); 80B rows keep 16B align
  __shared__ alignas(16) f16 As[2][128 * LDK];
  __shared__ alignas(16) f16 Bs[2][128 * LDK];  // stored transposed: [n][k]

  const int tid  = threadIdx.x;
  const int w    = tid >> 5;
  const int lane = tid & 31;
  const int half = lane >> 4;
  const int lrow = lane & 15;
  const int wm = w & 3;   // wave M coordinate (0..3)
  const int wn = w >> 2;  // wave N coordinate (0..1)
  const int bm0 = blockIdx.y * 128;
  const int bn0 = blockIdx.x * 128;

  v8f acc[2][4];
#pragma unroll
  for (int i = 0; i < 2; ++i)
#pragma unroll
    for (int j = 0; j < 4; ++j) acc[i][j] = v8f_zero();

  const int aRow = tid >> 1, aChunk = (tid & 1) * 16;   // A: 128 rows x 2 chunks
  const int bK = tid >> 3, bN0 = (tid & 7) * 16;        // B: 32 k x 8 n-chunks

  const f16* aBase = A + (size_t)(bm0 + aRow) * K + aChunk;
  const f16* bBase = Bw + (size_t)bK * Ncols + bn0 + bN0;
  const int nK = K >> 5;

  // ---- prologue: stage tile 0 into buffer 0 ----
  {
#if USE_ASYNC
    ASYNC_CP16(aBase,     &As[0][aRow * LDK + aChunk]);
    ASYNC_CP16(aBase + 8, &As[0][aRow * LDK + aChunk + 8]);
#else
    v8h a0 = *(const v8h*)aBase;
    v8h a1 = *(const v8h*)(aBase + 8);
    *(v8h*)&As[0][aRow * LDK + aChunk]     = a0;
    *(v8h*)&As[0][aRow * LDK + aChunk + 8] = a1;
#endif
    v8h b0 = *(const v8h*)bBase;
    v8h b1 = *(const v8h*)(bBase + 8);
#pragma unroll
    for (int i = 0; i < 8; ++i) Bs[0][(bN0 + i) * LDK + bK] = b0[i];
#pragma unroll
    for (int i = 0; i < 8; ++i) Bs[0][(bN0 + 8 + i) * LDK + bK] = b1[i];
  }
  ASYNC_WAIT();
  __syncthreads();

  int p = 0;
  for (int kb = 0; kb < nK; ++kb) {
    const int qb = p ^ 1;
    const bool more = (kb + 1) < nK;
    v8h nb0, nb1;
#if !USE_ASYNC
    v8h na0, na1;
#endif
    if (more) {
      const f16* an = aBase + (size_t)(kb + 1) * 32;
#if USE_ASYNC
      ASYNC_CP16(an,     &As[qb][aRow * LDK + aChunk]);
      ASYNC_CP16(an + 8, &As[qb][aRow * LDK + aChunk + 8]);
#else
      na0 = *(const v8h*)an;
      na1 = *(const v8h*)(an + 8);
#endif
      const f16* bn = bBase + (size_t)(kb + 1) * 32 * Ncols;
      nb0 = *(const v8h*)bn;
      nb1 = *(const v8h*)(bn + 8);
      if (kb + 2 < nK) {  // warm L2 two tiles ahead -> global_prefetch_b8
        __builtin_prefetch(aBase + (size_t)(kb + 2) * 32, 0, 3);
        __builtin_prefetch(bBase + (size_t)(kb + 2) * 32 * Ncols, 0, 3);
      }
    }

    // ---- compute from buffer p ----
    F16Frag af[2], bf[4];
#pragma unroll
    for (int i = 0; i < 2; ++i) {
      const f16* ab = &As[p][(wm * 32 + i * 16 + lrow) * LDK + half * 8];
      af[i].h[0] = *(const v8h*)ab;
      af[i].h[1] = *(const v8h*)(ab + 16);
    }
#pragma unroll
    for (int j = 0; j < 4; ++j) {
      const f16* bb = &Bs[p][(wn * 64 + j * 16 + lrow) * LDK + half * 8];
      bf[j].h[0] = *(const v8h*)bb;
      bf[j].h[1] = *(const v8h*)(bb + 16);
    }
#pragma unroll
    for (int i = 0; i < 2; ++i)
#pragma unroll
      for (int j = 0; j < 4; ++j)
        acc[i][j] = __builtin_amdgcn_wmma_f32_16x16x32_f16(
            false, af[i].v, false, bf[j].v, (short)0, acc[i][j], false, false);

    // ---- drain staged tile kb+1 into buffer qb ----
    if (more) {
#if !USE_ASYNC
      *(v8h*)&As[qb][aRow * LDK + aChunk]     = na0;
      *(v8h*)&As[qb][aRow * LDK + aChunk + 8] = na1;
#endif
#pragma unroll
      for (int i = 0; i < 8; ++i) Bs[qb][(bN0 + i) * LDK + bK] = nb0[i];
#pragma unroll
      for (int i = 0; i < 8; ++i) Bs[qb][(bN0 + 8 + i) * LDK + bK] = nb1[i];
    }
    ASYNC_WAIT();
    __syncthreads();
    p = qb;
  }

  // Epilogue. C layout: lane holds col n = lrow (+16*j...), rows half*8 + r.
#pragma unroll
  for (int i = 0; i < 2; ++i) {
#pragma unroll
    for (int j = 0; j < 4; ++j) {
#pragma unroll
      for (int r = 0; r < 8; ++r) {
        const int gm = bm0 + wm * 32 + i * 16 + half * 8 + r;
        const int gn = bn0 + wn * 64 + j * 16 + lrow;
        float val = acc[i][j][r];
        if constexpr (EPI == EPI_QKV) {
          const int three = gn / DIM;
          const int rem = gn - three * DIM;
          const int hh = rem >> 6, hd = rem & 63;
          const int bb = gm >> 11, nn = gm & 2047;  // N=2048
          f16* dst = (three == 0) ? qd : ((three == 1) ? kd : vd);
          dst[((size_t)(bb * NH + hh) * SEQ_N + nn) * HDIM + hd] = (f16)val;
        } else if constexpr (EPI == EPI_BIAS_RES) {
          const size_t idx = (size_t)gm * Ncols + gn;
          outf[idx] = val + bias[gn] + resid[idx];
        } else {  // EPI_BIAS_GELU
          const float t = val + bias[gn];
          const float ge = 0.5f * t * (1.0f + erff(t * 0.70710678118654752f));
          outh[(size_t)gm * Ncols + gn] = (f16)ge;
        }
      }
    }
  }
}

// ---------------------------------------------------------------------------
// Flash attention: one block per (64 Q rows, b*H+h). 128 threads = 4 waves,
// each wave owns 16 Q rows. Streams 64-key blocks with online softmax.
// Q/K staged via async-to-LDS; V transposed through registers.
// ---------------------------------------------------------------------------
__global__ __launch_bounds__(128) void attn_kernel(const f16* __restrict__ q,
                                                   const f16* __restrict__ k,
                                                   const f16* __restrict__ v,
                                                   f16* __restrict__ o) {
  constexpr int LD = 72;  // 144B row stride, 16B aligned
  __shared__ alignas(16) f16 Qs[64 * LD];
  __shared__ alignas(16) f16 Ks[64 * LD];
  __shared__ alignas(16) f16 Vt[64 * LD];          // transposed: [hd][key]
  __shared__ alignas(16) f16 Ps[4][16 * LD];       // per-wave P staging

  const int tid  = threadIdx.x;
  const int w    = tid >> 5;
  const int lane = tid & 31;
  const int half = lane >> 4;
  const int lrow = lane & 15;
  const int bh = blockIdx.y;          // 0..B*H-1
  const int q0 = blockIdx.x * 64;
  const size_t base = (size_t)bh * SEQ_N * HDIM;
  const f16* qg = q + base;
  const f16* kg = k + base;
  const f16* vg = v + base;

  const int sr = tid >> 1, sc0 = (tid & 1) * 32;   // staging coords

  // stage Q tile (64 x 64) once
  {
    const f16* src = qg + (size_t)(q0 + sr) * HDIM + sc0;
#if USE_ASYNC
#pragma unroll
    for (int c = 0; c < 4; ++c)
      ASYNC_CP16(src + c * 8, &Qs[sr * LD + sc0 + c * 8]);
#else
#pragma unroll
    for (int c = 0; c < 4; ++c)
      *(v8h*)&Qs[sr * LD + sc0 + c * 8] = *(const v8h*)(src + c * 8);
#endif
  }

  float m_run[8], l_run[8];
#pragma unroll
  for (int r = 0; r < 8; ++r) { m_run[r] = -1e30f; l_run[r] = 0.0f; }
  v8f oacc[4];
#pragma unroll
  for (int j = 0; j < 4; ++j) oacc[j] = v8f_zero();

  for (int kb = 0; kb < SEQ_N / 64; ++kb) {
    __syncthreads();
    // stage K block (async, row-major) and V block (register transpose)
    {
      const f16* ksrc = kg + (size_t)(kb * 64 + sr) * HDIM + sc0;
#if USE_ASYNC
#pragma unroll
      for (int c = 0; c < 4; ++c)
        ASYNC_CP16(ksrc + c * 8, &Ks[sr * LD + sc0 + c * 8]);
#else
#pragma unroll
      for (int c = 0; c < 4; ++c)
        *(v8h*)&Ks[sr * LD + sc0 + c * 8] = *(const v8h*)(ksrc + c * 8);
#endif
      const f16* vsrc = vg + (size_t)(kb * 64 + sr) * HDIM + sc0;
      v8h t[4];
#pragma unroll
      for (int c = 0; c < 4; ++c) t[c] = *(const v8h*)(vsrc + c * 8);
#pragma unroll
      for (int i = 0; i < 32; ++i) Vt[(sc0 + i) * LD + sr] = t[i >> 3][i & 7];
      if (kb + 1 < SEQ_N / 64) {  // warm L2 for the next key/value block
        __builtin_prefetch(kg + (size_t)((kb + 1) * 64 + sr) * HDIM + sc0, 0, 3);
        __builtin_prefetch(vg + (size_t)((kb + 1) * 64 + sr) * HDIM + sc0, 0, 3);
      }
    }
    ASYNC_WAIT();
    __syncthreads();

    // S = Q * K^T  (per wave: 16 x 64)
    v8f s[4];
#pragma unroll
    for (int j = 0; j < 4; ++j) s[j] = v8f_zero();
#pragma unroll
    for (int ks = 0; ks < 2; ++ks) {
      F16Frag aq;
      const f16* ab = &Qs[(w * 16 + lrow) * LD + ks * 32 + half * 8];
      aq.h[0] = *(const v8h*)ab;
      aq.h[1] = *(const v8h*)(ab + 16);
#pragma unroll
      for (int j = 0; j < 4; ++j) {
        F16Frag bk;
        const f16* bb = &Ks[(j * 16 + lrow) * LD + ks * 32 + half * 8];
        bk.h[0] = *(const v8h*)bb;
        bk.h[1] = *(const v8h*)(bb + 16);
        s[j] = __builtin_amdgcn_wmma_f32_16x16x32_f16(
            false, aq.v, false, bk.v, (short)0, s[j], false, false);
      }
    }

    // online softmax per row (rows live in 16-lane halves)
#pragma unroll
    for (int r = 0; r < 8; ++r) {
      float e[4];
#pragma unroll
      for (int j = 0; j < 4; ++j) e[j] = s[j][r] * ATT_SCALE;
      float rm = fmaxf(fmaxf(e[0], e[1]), fmaxf(e[2], e[3]));
#pragma unroll
      for (int m = 1; m <= 8; m <<= 1) rm = fmaxf(rm, __shfl_xor(rm, m, 32));
      const float mn = fmaxf(m_run[r], rm);
      const float al = __expf(m_run[r] - mn);
      float rsum = 0.0f;
      float pv[4];
#pragma unroll
      for (int j = 0; j < 4; ++j) { pv[j] = __expf(e[j] - mn); rsum += pv[j]; }
#pragma unroll
      for (int m = 1; m <= 8; m <<= 1) rsum += __shfl_xor(rsum, m, 32);
      l_run[r] = l_run[r] * al + rsum;
      m_run[r] = mn;
#pragma unroll
      for (int j = 0; j < 4; ++j) oacc[j][r] *= al;
#pragma unroll
      for (int j = 0; j < 4; ++j)
        Ps[w][(half * 8 + r) * LD + j * 16 + lrow] = (f16)pv[j];
    }

    // O += P * V
#pragma unroll
    for (int ks = 0; ks < 2; ++ks) {
      F16Frag ap;
      const f16* ab = &Ps[w][lrow * LD + ks * 32 + half * 8];
      ap.h[0] = *(const v8h*)ab;
      ap.h[1] = *(const v8h*)(ab + 16);
#pragma unroll
      for (int j = 0; j < 4; ++j) {
        F16Frag bv;
        const f16* bb = &Vt[(j * 16 + lrow) * LD + ks * 32 + half * 8];
        bv.h[0] = *(const v8h*)bb;
        bv.h[1] = *(const v8h*)(bb + 16);
        oacc[j] = __builtin_amdgcn_wmma_f32_16x16x32_f16(
            false, ap.v, false, bv.v, (short)0, oacc[j], false, false);
      }
    }
  }

  // write O normalized, back to token-major [B*N, D] at this head's columns
  const int bIdx = bh / NH, hIdx = bh % NH;
#pragma unroll
  for (int j = 0; j < 4; ++j) {
#pragma unroll
    for (int r = 0; r < 8; ++r) {
      const float val = oacc[j][r] / l_run[r];
      const int n = q0 + w * 16 + half * 8 + r;
      o[(size_t)(bIdx * SEQ_N + n) * DIM + hIdx * HDIM + j * 16 + lrow] = (f16)val;
    }
  }
}

// ---------------------------------------------------------------------------
// Host launcher
// ---------------------------------------------------------------------------
extern "C" void kernel_launch(void* const* d_in, const int* in_sizes, int n_in,
                              void* d_out, int out_size, void* d_ws, size_t ws_size,
                              hipStream_t stream) {
  (void)in_sizes; (void)n_in; (void)out_size; (void)ws_size;
  const float* x      = (const float*)d_in[0];
  const float* ln1_g  = (const float*)d_in[1];
  const float* ln1_b  = (const float*)d_in[2];
  const float* qkv_w  = (const float*)d_in[3];
  const float* proj_w = (const float*)d_in[4];
  const float* proj_b = (const float*)d_in[5];
  const float* ln2_g  = (const float*)d_in[6];
  const float* ln2_b  = (const float*)d_in[7];
  const float* fc1_w  = (const float*)d_in[8];
  const float* fc1_b  = (const float*)d_in[9];
  const float* fc2_w  = (const float*)d_in[10];
  const float* fc2_b  = (const float*)d_in[11];
  float* out = (float*)d_out;

  char* ws = (char*)d_ws;
  size_t off = 0;
  auto carve = [&](size_t bytes) -> void* {
    void* p = ws + off;
    off = (off + bytes + 255) & ~(size_t)255;
    return p;
  };

  f16* wq = (f16*)carve((size_t)DIM * 3 * DIM * sizeof(f16));
  f16* wp = (f16*)carve((size_t)DIM * DIM * sizeof(f16));
  f16* w1 = (f16*)carve((size_t)DIM * HID * sizeof(f16));
  f16* w2 = (f16*)carve((size_t)HID * DIM * sizeof(f16));
  f16* h1 = (f16*)carve((size_t)MTOT * DIM * sizeof(f16));
  f16* qd = (f16*)carve((size_t)MTOT * DIM * sizeof(f16));
  f16* kd = (f16*)carve((size_t)MTOT * DIM * sizeof(f16));
  f16* vd = (f16*)carve((size_t)MTOT * DIM * sizeof(f16));
  f16* od = (f16*)carve((size_t)MTOT * DIM * sizeof(f16));
  float* x1 = (float*)carve((size_t)MTOT * DIM * sizeof(float));
  f16* h2 = (f16*)carve((size_t)MTOT * DIM * sizeof(f16));
  f16* g1 = (f16*)carve((size_t)MTOT * HID * sizeof(f16));

  // 1) weights -> f16
  cvt_f32_to_f16<<<2048, 256, 0, stream>>>(qkv_w, wq, DIM * 3 * DIM);
  cvt_f32_to_f16<<<2048, 256, 0, stream>>>(proj_w, wp, DIM * DIM);
  cvt_f32_to_f16<<<2048, 256, 0, stream>>>(fc1_w, w1, DIM * HID);
  cvt_f32_to_f16<<<2048, 256, 0, stream>>>(fc2_w, w2, HID * DIM);

  // 2) LN1
  ln_to_f16<<<MTOT, 256, 0, stream>>>(x, ln1_g, ln1_b, h1);

  // 3) QKV GEMM (split to head-major q/k/v)
  gemm_f16<EPI_QKV><<<dim3((3 * DIM) / 128, MTOT / 128), 256, 0, stream>>>(
      h1, wq, MTOT, 3 * DIM, DIM, nullptr, nullptr, nullptr, nullptr, qd, kd, vd);

  // 4) flash attention
  attn_kernel<<<dim3(SEQ_N / 64, 4 * NH), 128, 0, stream>>>(qd, kd, vd, od);

  // 5) proj GEMM + bias + residual(x) -> x1 (f32)
  gemm_f16<EPI_BIAS_RES><<<dim3(DIM / 128, MTOT / 128), 256, 0, stream>>>(
      od, wp, MTOT, DIM, DIM, proj_b, x, x1, nullptr, nullptr, nullptr, nullptr);

  // 6) LN2
  ln_to_f16<<<MTOT, 256, 0, stream>>>(x1, ln2_g, ln2_b, h2);

  // 7) FC1 GEMM + bias + exact GELU -> g1 (f16)
  gemm_f16<EPI_BIAS_GELU><<<dim3(HID / 128, MTOT / 128), 256, 0, stream>>>(
      h2, w1, MTOT, HID, DIM, fc1_b, nullptr, nullptr, g1, nullptr, nullptr, nullptr);

  // 8) FC2 GEMM + bias + residual(x1) -> out (f32)
  gemm_f16<EPI_BIAS_RES><<<dim3(DIM / 128, MTOT / 128), 256, 0, stream>>>(
      g1, w2, MTOT, DIM, HID, fc2_b, x1, out, nullptr, nullptr, nullptr, nullptr);
}
